// GaugeConvolutionBlock_71923522339031
// MI455X (gfx1250) — compile-verified, matrix operand order checked by reading, not run on hardware
//
#include <hip/hip_runtime.h>
#include <hip/hip_bf16.h>
#include <math.h>

// Problem constants
#define BB   4
#define SS   2048
#define DD   128
#define DFF  512
#define RR   8
#define WW   8
#define EPS  1e-5f

typedef __attribute__((ext_vector_type(16))) _Float16 v16h;
typedef __attribute__((ext_vector_type(8)))  float    v8f;

union FragU { float4 f[2]; v16h h; };

// ---------------------------------------------------------------------------
// Fragment loaders.  Assumed wave32 WMMA 16x16x32 f16 layout (ISA 7.12.2):
//   A (16x32, MxK): lane L (group g=L>>4) holds row m=L&15,
//                   halves 0..7  -> K = g*8+0..7
//                   halves 8..15 -> K = 16+g*8+0..7
//   B (32x16, KxN): symmetric with m -> n.
//   C (16x16): VGPR j holds M=j (lanes 0-15) / M=8+j (lanes 16-31), N=lane&15.
// ---------------------------------------------------------------------------
static __device__ __forceinline__ v16h load_frag_a(const _Float16* base, int stride,
                                                   int mBase, int kBase, int lane) {
    int m = mBase + (lane & 15);
    int g = (lane >> 4) & 1;
    const _Float16* p0 = base + (size_t)m * stride + kBase + g * 8;
    FragU u;
    u.f[0] = *(const float4*)p0;        // K = kBase+g*8 .. +7      (16B)
    u.f[1] = *(const float4*)(p0 + 16); // K = kBase+16+g*8 .. +7   (16B)
    return u.h;
}

static __device__ __forceinline__ v16h load_frag_b(const _Float16* packed,
                                                   size_t fragIdx, int lane) {
    const _Float16* p = packed + (fragIdx * 32 + (size_t)lane) * 16;
    FragU u;
    u.f[0] = *(const float4*)p;
    u.f[1] = *(const float4*)(p + 8);
    return u.h;
}

static __device__ __forceinline__ v8f wmma_f16(v16h a, v16h b, v8f c) {
    return __builtin_amdgcn_wmma_f32_16x16x32_f16(false, a, false, b, (short)0, c,
                                                  false, false);
}

static __device__ __forceinline__ float gelu_exact(float x) {
    return 0.5f * x * (1.0f + erff(x * 0.70710678118654752f));
}

// ---------------------------------------------------------------------------
// Pack a row-major KxN f32 weight matrix into per-fragment f16 layout:
// frag (kt,nt) occupies 512 contiguous halves; lane L owns halves [L*16, L*16+16).
// ---------------------------------------------------------------------------
__global__ void pack_b_kernel(const float* __restrict__ src, _Float16* __restrict__ dst,
                              int K, int N) {
    int frag = blockIdx.x;
    int lane = threadIdx.x;           // 32 threads
    int nTiles = N >> 4;
    int kt = frag / nTiles, nt = frag % nTiles;
    int n = (nt << 4) + (lane & 15);
    int g = (lane >> 4) & 1;
    _Float16* out = dst + ((size_t)frag * 32 + lane) * 16;
#pragma unroll
    for (int i = 0; i < 16; ++i) {
        int k = (kt << 5) + ((i < 8) ? (g * 8 + i) : (16 + g * 8 + (i - 8)));
        out[i] = (_Float16)src[(size_t)k * N + n];
    }
}

// ---------------------------------------------------------------------------
// Spectral norm of Wg2 (512 x 2048): 10 power iterations, one 512-thread block.
// ---------------------------------------------------------------------------
__global__ __launch_bounds__(512) void spectral_kernel(const float* __restrict__ Wg2,
                                                       float* __restrict__ inv_sigma) {
    __shared__ float ush[2048];
    __shared__ float vsh[512];
    __shared__ float red[512];
    int t = threadIdx.x;

    for (int j = t; j < 2048; j += 512) ush[j] = 0.022097086912079608f; // 1/sqrt(2048)
    __syncthreads();

    const float* wrow = Wg2 + (size_t)t * 2048;
    for (int it = 0; it < 10; ++it) {
        // v = normalize(Wg2 @ u)
        float acc = 0.f;
        for (int k = 0; k < 2048; ++k) acc += wrow[k] * ush[k];
        red[t] = acc * acc;
        __syncthreads();
        for (int sr = 256; sr > 0; sr >>= 1) {
            if (t < sr) red[t] += red[t + sr];
            __syncthreads();
        }
        float vn = 1.f / (sqrtf(red[0]) + 1e-12f);
        __syncthreads();
        vsh[t] = acc * vn;
        __syncthreads();

        // u = normalize(Wg2^T @ v)
        float a4[4];
        float ss = 0.f;
#pragma unroll
        for (int jj = 0; jj < 4; ++jj) {
            int j = t + jj * 512;
            float a = 0.f;
            for (int i = 0; i < 512; ++i) a += Wg2[(size_t)i * 2048 + j] * vsh[i];
            a4[jj] = a;
            ss += a * a;
        }
        red[t] = ss;
        __syncthreads();
        for (int sr = 256; sr > 0; sr >>= 1) {
            if (t < sr) red[t] += red[t + sr];
            __syncthreads();
        }
        float un = 1.f / (sqrtf(red[0]) + 1e-12f);
        __syncthreads();
#pragma unroll
        for (int jj = 0; jj < 4; ++jj) ush[t + jj * 512] = a4[jj] * un;
        __syncthreads();
    }
    // sigma = v . (Wg2 @ u)
    float acc = 0.f;
    for (int k = 0; k < 2048; ++k) acc += wrow[k] * ush[k];
    red[t] = vsh[t] * acc;
    __syncthreads();
    for (int sr = 256; sr > 0; sr >>= 1) {
        if (t < sr) red[t] += red[t + sr];
        __syncthreads();
    }
    if (t == 0) inv_sigma[0] = 1.f / red[0];
}

// ---------------------------------------------------------------------------
// LayerNorm 1: one block per (b,s) row of 128. Writes phi as f16 (only consumer).
// ---------------------------------------------------------------------------
__global__ __launch_bounds__(128) void ln1_kernel(const float* __restrict__ q,
                                                  const float* __restrict__ g,
                                                  const float* __restrict__ bta,
                                                  _Float16* __restrict__ phi16) {
    __shared__ float red[128];
    size_t base = (size_t)blockIdx.x * DD;
    int t = threadIdx.x;
    float x = q[base + t];
    red[t] = x;
    __syncthreads();
    for (int sr = 64; sr > 0; sr >>= 1) {
        if (t < sr) red[t] += red[t + sr];
        __syncthreads();
    }
    float mu = red[0] * (1.f / DD);
    __syncthreads();
    float dx = x - mu;
    red[t] = dx * dx;
    __syncthreads();
    for (int sr = 64; sr > 0; sr >>= 1) {
        if (t < sr) red[t] += red[t + sr];
        __syncthreads();
    }
    float var = red[0] * (1.f / DD);
    phi16[base + t] = (_Float16)(dx * rsqrtf(var + EPS) * g[t] + bta[t]);
}

// ---------------------------------------------------------------------------
// Fused gauge kernel. One block = 8 s-positions x 8 windows = 64 pair-rows.
//   LDS: Ash pairs (64x256 f16), Hsh h (64x512 f16), Tsh t (64x8 f32),
//        LRsh lowrank (64x128 f32)  => 130 KB.
// ---------------------------------------------------------------------------
__global__ __launch_bounds__(256) void gauge_kernel(
    const _Float16* __restrict__ phi16, const float* __restrict__ q,
    const _Float16* __restrict__ Wg1p, const float* __restrict__ bg1,
    const _Float16* __restrict__ Wg2p, const float* __restrict__ bg2,
    const float* __restrict__ inv_sigma_p, const float* __restrict__ gauge_scale_p,
    float* __restrict__ q2) {
    __shared__ __align__(16) _Float16 Ash[64 * 256];
    __shared__ __align__(16) _Float16 Hsh[64 * 512];
    __shared__ float Tsh[64 * 8];
    __shared__ float LRsh[64 * 128];

    int blk = blockIdx.x;
    int b = blk >> 8;            // 256 s-tiles per batch
    int s0 = (blk & 255) << 3;   // 8 s per tile
    int tid = threadIdx.x;
    int wave = tid >> 5;
    int lane = tid & 31;

    // ---- build pairs tile with 16B (b128) copies:
    //   row = s_local*8 + w; seg 0 = phi_i (cols 0..127), seg 1 = phi_win.
    //   2048 float4 transfers, 8 per thread.
    for (int idx = tid; idx < 2048; idx += 256) {
        int row = idx >> 5;          // 0..63
        int seg = (idx >> 4) & 1;    // 0 or 1
        int chunk = idx & 15;        // 16B chunk within the 256B segment
        int s_local = row >> 3;
        int w = row & 7;
        int s = s0 + s_local;
        float4 val;
        if (seg == 0) {
            val = *(const float4*)(phi16 + ((size_t)(b * SS + s)) * DD + chunk * 8);
        } else {
            int sw = s - 3 + w;
            if (sw >= 0 && sw < SS)
                val = *(const float4*)(phi16 + ((size_t)(b * SS + sw)) * DD + chunk * 8);
            else
                val = make_float4(0.f, 0.f, 0.f, 0.f);
        }
        *(float4*)(Ash + (size_t)row * 256 + seg * 128 + chunk * 8) = val;
    }
    for (int i = tid; i < 64 * 8; i += 256) Tsh[i] = 0.f;
    for (int i = tid; i < 64 * 128; i += 256) LRsh[i] = 0.f;
    __syncthreads();

    // ---- GEMM1: (64x256) @ Wg1 (256x512), GELU, store f16 h. Tiles: 4m x 32n.
    for (int tIdx = wave; tIdx < 128; tIdx += 8) {
        int mt = tIdx & 3;
        int nt = tIdx >> 2;
        v8f acc = {};
#pragma unroll
        for (int kt = 0; kt < 8; ++kt) {
            v16h a = load_frag_a(Ash, 256, mt << 4, kt << 5, lane);
            v16h bf = load_frag_b(Wg1p, (size_t)kt * 32 + nt, lane);
            acc = wmma_f16(a, bf, acc);
        }
        int n = (nt << 4) + (lane & 15);
        int mBase = (mt << 4) + ((lane & 16) ? 8 : 0);
        float bias = bg1[n];
#pragma unroll
        for (int j = 0; j < 8; ++j) {
            float x = acc[j] + bias;
            Hsh[(size_t)(mBase + j) * 512 + n] = (_Float16)gelu_exact(x);
        }
    }
    __syncthreads();

    float inv_sigma = inv_sigma_p[0];

    // ---- GEMM2 pass 1: v-columns (1024..2047). t[row][r] += f * phi_win[row][d]
    for (int tIdx = wave; tIdx < 256; tIdx += 8) {
        int mt = tIdx & 3;
        int nt = 64 + (tIdx >> 2);   // global n-tile in [64,128)
        // warm GL2/WGP$ for the next tile's B fragments
        if (tIdx + 8 < 256)
            __builtin_prefetch(Wg2p + ((size_t)(64 + ((tIdx + 8) >> 2))) * 32 * 16, 0, 1);
        v8f acc = {};
#pragma unroll
        for (int kt = 0; kt < 16; ++kt) {
            v16h a = load_frag_a(Hsh, 512, mt << 4, kt << 5, lane);
            v16h bf = load_frag_b(Wg2p, (size_t)kt * 128 + nt, lane);
            acc = wmma_f16(a, bf, acc);
        }
        int col = (nt << 4) + (lane & 15);   // 1024..2047
        int vcol = col - 1024;
        int d = vcol >> 3;
        int r = vcol & 7;
        float bias = bg2[col];
        int mBase = (mt << 4) + ((lane & 16) ? 8 : 0);
#pragma unroll
        for (int j = 0; j < 8; ++j) {
            int m = mBase + j;
            float f = acc[j] * inv_sigma + bias;
            float pw = (float)Ash[(size_t)m * 256 + 128 + d];
            atomicAdd(&Tsh[m * 8 + r], f * pw);
        }
    }
    __syncthreads();

    // ---- GEMM2 pass 2: u-columns (0..1023). lowrank[row][d] += f * t[row][r]
    for (int tIdx = wave; tIdx < 256; tIdx += 8) {
        int mt = tIdx & 3;
        int nt = tIdx >> 2;          // global n-tile in [0,64)
        if (tIdx + 8 < 256)
            __builtin_prefetch(Wg2p + ((size_t)((tIdx + 8) >> 2)) * 32 * 16, 0, 1);
        v8f acc = {};
#pragma unroll
        for (int kt = 0; kt < 16; ++kt) {
            v16h a = load_frag_a(Hsh, 512, mt << 4, kt << 5, lane);
            v16h bf = load_frag_b(Wg2p, (size_t)kt * 128 + nt, lane);
            acc = wmma_f16(a, bf, acc);
        }
        int col = (nt << 4) + (lane & 15);
        int d = col >> 3;
        int r = col & 7;
        float bias = bg2[col];
        int mBase = (mt << 4) + ((lane & 16) ? 8 : 0);
#pragma unroll
        for (int j = 0; j < 8; ++j) {
            int m = mBase + j;
            float f = acc[j] * inv_sigma + bias;
            atomicAdd(&LRsh[m * 128 + d], f * Tsh[m * 8 + r]);
        }
    }
    __syncthreads();

    // ---- aggregate over windows, residual into q2
    float gsc = gauge_scale_p[0];
    for (int idx = tid; idx < 8 * 128; idx += 256) {
        int s_local = idx >> 7;
        int dd = idx & 127;
        float a = 0.f;
#pragma unroll
        for (int w = 0; w < WW; ++w) {
            int row = (s_local << 3) + w;
            a += 0.98f * (float)Ash[(size_t)row * 256 + 128 + dd] +
                 0.02f * LRsh[row * 128 + dd];
        }
        size_t gi = ((size_t)(b * SS + s0 + s_local)) * DD + dd;
        q2[gi] = q[gi] + gsc * a;
    }
}

// ---------------------------------------------------------------------------
// MLP kernel: LN2 + 128->512 GELU -> 128, residual. 64 rows per block.
// ---------------------------------------------------------------------------
__global__ __launch_bounds__(256) void mlp_kernel(
    const float* __restrict__ q2, const float* __restrict__ n2g,
    const float* __restrict__ n2b, const _Float16* __restrict__ W1p,
    const float* __restrict__ b1, const _Float16* __restrict__ W2p,
    const float* __restrict__ b2, const float* __restrict__ mlp_scale_p,
    float* __restrict__ out) {
    __shared__ __align__(16) _Float16 Psh[64 * 128];
    __shared__ __align__(16) _Float16 Hsh[64 * 512];
    __shared__ float redS[256];
    __shared__ float redQ[256];
    __shared__ float stat[64 * 2];    // mu, rsigma per row
    int tid = threadIdx.x;
    int wave = tid >> 5;
    int lane = tid & 31;
    size_t row0 = (size_t)blockIdx.x * 64;

    // LayerNorm 2: 4 threads per row
    {
        int row = tid >> 2;
        int part = tid & 3;
        const float* x = q2 + (row0 + row) * DD;
        float s = 0.f, ssq = 0.f;
        for (int k = part * 32; k < part * 32 + 32; ++k) {
            float v = x[k];
            s += v;
            ssq += v * v;
        }
        redS[tid] = s;
        redQ[tid] = ssq;
        __syncthreads();
        if (part == 0) {
            float st = redS[tid] + redS[tid + 1] + redS[tid + 2] + redS[tid + 3];
            float sq = redQ[tid] + redQ[tid + 1] + redQ[tid + 2] + redQ[tid + 3];
            float mu = st * (1.f / DD);
            float var = sq * (1.f / DD) - mu * mu;
            stat[row * 2] = mu;
            stat[row * 2 + 1] = rsqrtf(var + EPS);
        }
        __syncthreads();
        float mu = stat[row * 2];
        float rs = stat[row * 2 + 1];
        for (int k = part * 32; k < part * 32 + 32; ++k)
            Psh[row * DD + k] = (_Float16)((x[k] - mu) * rs * n2g[k] + n2b[k]);
    }
    __syncthreads();

    // GEMM1: (64x128) @ W1 (128x512), GELU. Tiles 4m x 32n.
    for (int tIdx = wave; tIdx < 128; tIdx += 8) {
        int mt = tIdx & 3;
        int nt = tIdx >> 2;
        v8f acc = {};
#pragma unroll
        for (int kt = 0; kt < 4; ++kt) {
            v16h a = load_frag_a(Psh, 128, mt << 4, kt << 5, lane);
            v16h bf = load_frag_b(W1p, (size_t)kt * 32 + nt, lane);
            acc = wmma_f16(a, bf, acc);
        }
        int n = (nt << 4) + (lane & 15);
        int mBase = (mt << 4) + ((lane & 16) ? 8 : 0);
        float bias = b1[n];
#pragma unroll
        for (int j = 0; j < 8; ++j) {
            float x = acc[j] + bias;
            Hsh[(size_t)(mBase + j) * 512 + n] = (_Float16)gelu_exact(x);
        }
    }
    __syncthreads();

    // GEMM2: (64x512) @ W2 (512x128), residual. Tiles 4m x 8n.
    float msc = mlp_scale_p[0];
    for (int tIdx = wave; tIdx < 32; tIdx += 8) {
        int mt = tIdx & 3;
        int nt = tIdx >> 2;
        v8f acc = {};
#pragma unroll
        for (int kt = 0; kt < 16; ++kt) {
            v16h a = load_frag_a(Hsh, 512, mt << 4, kt << 5, lane);
            v16h bf = load_frag_b(W2p, (size_t)kt * 8 + nt, lane);
            acc = wmma_f16(a, bf, acc);
        }
        int n = (nt << 4) + (lane & 15);
        int mBase = (mt << 4) + ((lane & 16) ? 8 : 0);
        float bias = b2[n];
#pragma unroll
        for (int j = 0; j < 8; ++j) {
            size_t gi = (row0 + mBase + j) * DD + n;
            out[gi] = q2[gi] + msc * (acc[j] + bias);
        }
    }
}

// ---------------------------------------------------------------------------
extern "C" void kernel_launch(void* const* d_in, const int* in_sizes, int n_in,
                              void* d_out, int out_size, void* d_ws, size_t ws_size,
                              hipStream_t stream) {
    const float* q   = (const float*)d_in[0];
    const float* n1g = (const float*)d_in[1];
    const float* n1b = (const float*)d_in[2];
    const float* n2g = (const float*)d_in[3];
    const float* n2b = (const float*)d_in[4];
    const float* Wg1 = (const float*)d_in[5];   // 256 x 512
    const float* bg1 = (const float*)d_in[6];
    const float* Wg2 = (const float*)d_in[7];   // 512 x 2048
    const float* bg2 = (const float*)d_in[8];
    const float* W1  = (const float*)d_in[9];   // 128 x 512
    const float* b1  = (const float*)d_in[10];
    const float* W2  = (const float*)d_in[11];  // 512 x 128
    const float* b2  = (const float*)d_in[12];
    const float* gauge_scale = (const float*)d_in[13];
    const float* mlp_scale   = (const float*)d_in[14];
    float* out = (float*)d_out;

    // Workspace carve-up (~8.9 MB total)
    char* w = (char*)d_ws;
    float*     inv_sigma = (float*)w;                     // 256 B slot
    _Float16*  phi16 = (_Float16*)(w + 256);              // 2 MB
    float*     q2    = (float*)(w + 256 + 2097152);       // 4 MB
    char* wp = w + 256 + 2097152 + 4194304;
    _Float16* Wg1p = (_Float16*)wp;        wp += 262144;  // 256x512 f16
    _Float16* Wg2p = (_Float16*)wp;        wp += 2097152; // 512x2048 f16
    _Float16* W1p  = (_Float16*)wp;        wp += 131072;  // 128x512 f16
    _Float16* W2p  = (_Float16*)wp;                       // 512x128 f16

    pack_b_kernel<<<(256 / 32) * (512 / 16), 32, 0, stream>>>(Wg1, Wg1p, 256, 512);
    pack_b_kernel<<<(512 / 32) * (2048 / 16), 32, 0, stream>>>(Wg2, Wg2p, 512, 2048);
    pack_b_kernel<<<(128 / 32) * (512 / 16), 32, 0, stream>>>(W1, W1p, 128, 512);
    pack_b_kernel<<<(512 / 32) * (128 / 16), 32, 0, stream>>>(W2, W2p, 512, 128);

    spectral_kernel<<<1, 512, 0, stream>>>(Wg2, inv_sigma);

    ln1_kernel<<<BB * SS, 128, 0, stream>>>(q, n1g, n1b, phi16);

    gauge_kernel<<<BB * (SS / 8), 256, 0, stream>>>(phi16, q, Wg1p, bg1, Wg2p, bg2,
                                                    inv_sigma, gauge_scale, q2);

    mlp_kernel<<<(BB * SS) / 64, 256, 0, stream>>>(q2, n2g, n2b, W1p, b1, W2p, b2,
                                                   mlp_scale, out);
}
